// BiDisen_4071628997319
// MI455X (gfx1250) — matrix operand order repaired and checked by reading, not compiled, not run on hardware
//
#include <hip/hip_runtime.h>
#include <hip/hip_bf16.h>
#include <math.h>

#define MLEN   6000
#define NLEN   6000
#define DT     128
#define M_NB   20
#define NTOT   12000
#define RPAIR  100000
#define ACOEF  0.9f
#define BCOEF  0.01f
#define NSTRIP 375   // 6000 / 16

typedef __attribute__((ext_vector_type(16))) __bf16 v16bf;
typedef __attribute__((ext_vector_type(8)))  float  v8f;

static __device__ __forceinline__ float wave_red_sum(float v) {
    v += __shfl_xor(v, 1, 32);
    v += __shfl_xor(v, 2, 32);
    v += __shfl_xor(v, 4, 32);
    v += __shfl_xor(v, 8, 32);
    v += __shfl_xor(v, 16, 32);
    return v;
}

// sigmoid with single v_rcp_f32 instead of the full IEEE division sequence
static __device__ __forceinline__ float fast_sigmoid(float g) {
    return __builtin_amdgcn_rcpf(1.f + __expf(-g));
}

// ---------------------------------------------------------------------------
// Routing conv: one wave (32 lanes) per node. Lane l owns elements 4l..4l+3
// of the 128-wide row (each k-group of 16 elems = 4 consecutive lanes).
// Two sweeps over the 20 neighbors: online-softmax stats, then fused
// attention-weighted accumulation.
// ---------------------------------------------------------------------------
__global__ void routing_kernel(const float* __restrict__ xu,
                               const float* __restrict__ xv,
                               const int*   __restrict__ nbm,
                               const int*   __restrict__ nbn,
                               const float* __restrict__ att,
                               float*       __restrict__ out)
{
    const int wid  = threadIdx.x >> 5;
    const int lane = threadIdx.x & 31;
    const int node = blockIdx.x * (blockDim.x >> 5) + wid;
    if (node >= NTOT) return;

    const float4 att0 = *(const float4*)(att + 4 * lane);        // att[:128]
    const float4 att1 = *(const float4*)(att + DT + 4 * lane);   // att[128:]

    const float* xrow = (node < MLEN) ? (xu + (size_t)node * DT)
                                      : (xv + (size_t)(node - MLEN) * DT);
    const float4 x4 = *(const float4*)(xrow + 4 * lane);
    const float ex = wave_red_sum(x4.x*att0.x + x4.y*att0.y + x4.z*att0.z + x4.w*att0.w);

    // pass 1: online softmax over the 20 neighbor scores
    float mx = -3.0e38f, ssum = 0.f;
    for (int m = 0; m < M_NB; ++m) {
        const int idx = (node < MLEN) ? nbm[node * M_NB + m]
                                      : (nbn[(node - MLEN) * M_NB + m] + MLEN);
        const float* zr = (idx < MLEN) ? (xu + (size_t)idx * DT)
                                       : (xv + (size_t)(idx - MLEN) * DT);
        const float4 z4 = *(const float4*)(zr + 4 * lane);
        const float e = ex + wave_red_sum(z4.x*att1.x + z4.y*att1.y + z4.z*att1.z + z4.w*att1.w);
        const float nm = fmaxf(mx, e);
        ssum = ssum * __expf(mx - nm) + __expf(e - nm);
        mx = nm;
    }
    const float inv_ssum = __builtin_amdgcn_rcpf(ssum);

    // pass 2: u = A * sum_m z * (attn_m * zsum_k) + x
    float4 acc = make_float4(0.f, 0.f, 0.f, 0.f);
    for (int m = 0; m < M_NB; ++m) {
        const int idx = (node < MLEN) ? nbm[node * M_NB + m]
                                      : (nbn[(node - MLEN) * M_NB + m] + MLEN);
        const float* zr = (idx < MLEN) ? (xu + (size_t)idx * DT)
                                       : (xv + (size_t)(idx - MLEN) * DT);
        const float4 z4 = *(const float4*)(zr + 4 * lane);
        const float e = ex + wave_red_sum(z4.x*att1.x + z4.y*att1.y + z4.z*att1.z + z4.w*att1.w);
        const float attn = __expf(e - mx) * inv_ssum;
        // zsum over this lane's k-group (4 lanes of 4 elems = dd=16)
        float ps = z4.x + z4.y + z4.z + z4.w;
        ps += __shfl_xor(ps, 1, 32);
        ps += __shfl_xor(ps, 2, 32);
        const float w = attn * ps;
        acc.x += z4.x * w; acc.y += z4.y * w; acc.z += z4.z * w; acc.w += z4.w * w;
    }
    float4 o;
    o.x = ACOEF * acc.x + x4.x;
    o.y = ACOEF * acc.y + x4.y;
    o.z = ACOEF * acc.z + x4.z;
    o.w = ACOEF * acc.w + x4.w;
    *(float4*)(out + (size_t)node * DT + 4 * lane) = o;
}

// ---------------------------------------------------------------------------
// fp32 -> bf16 copy of embeddings (WMMA operand matrix, 3 MB => L2 resident)
// ---------------------------------------------------------------------------
__global__ void cvt_bf16_kernel(const float* __restrict__ src,
                                __bf16* __restrict__ dst, int n)
{
    const int i = blockIdx.x * blockDim.x + threadIdx.x;
    if (i < n) dst[i] = (__bf16)src[i];
}

// ---------------------------------------------------------------------------
// Per-row inverse rowsum of the 6000x6000 target matrix (streamed NT).
// ---------------------------------------------------------------------------
__global__ void rowinv_kernel(const float* __restrict__ P,
                              float* __restrict__ invrs, int n)
{
    __shared__ float sh[256];
    const int row = blockIdx.x;
    const float* pr = P + (size_t)row * n;
    float s = 0.f;
    for (int j = threadIdx.x; j < n; j += blockDim.x)
        s += __builtin_nontemporal_load(pr + j);
    sh[threadIdx.x] = s;
    __syncthreads();
    for (int o = 128; o > 0; o >>= 1) {
        if (threadIdx.x < (unsigned)o) sh[threadIdx.x] += sh[threadIdx.x + o];
        __syncthreads();
    }
    if (threadIdx.x == 0) invrs[row] = 1.f / sh[0];
}

// ---------------------------------------------------------------------------
// Gram-matrix fragments. One wave owns a 16-row strip of X (6000x128 bf16).
// A layout (ISA 7.12.2, 16-bit A 16x32): lane<16 -> row M=lane, K {0..7,16..23};
// lane>=16 -> row M=lane-16, K {8..15,24..31} of each K=32 chunk.
// B layout (32x16 col-striped): lane<16 -> col N=lane, K {0..7,16..23}; lane>=16 -> K {8..15,24..31}.
// ---------------------------------------------------------------------------
static __device__ __forceinline__ void load_a_frags(const __bf16* __restrict__ X,
                                                    int i0, int lh, int hi,
                                                    v16bf afrag[4])
{
    const __bf16* rowp = X + (size_t)(i0 + lh) * DT;
#pragma unroll
    for (int kc = 0; kc < 4; ++kc) {
        const int k0 = kc * 32 + hi * 8;
        v16bf a;
#pragma unroll
        for (int t = 0; t < 8; ++t) a[t] = rowp[k0 + t];
#pragma unroll
        for (int t = 0; t < 8; ++t) a[8 + t] = rowp[k0 + 16 + t];
        afrag[kc] = a;
    }
}

static __device__ __forceinline__ void load_b_frags(const __bf16* __restrict__ X,
                                                    int j0, int lh, int hi,
                                                    v16bf bfrag[4])
{
    const __bf16* rowq = X + (size_t)(j0 + lh) * DT;
#pragma unroll
    for (int kc = 0; kc < 4; ++kc) {
        const int kb = kc * 32 + hi * 8;
        v16bf b;
#pragma unroll
        for (int t = 0; t < 8; ++t) b[t] = rowq[kb + t];
#pragma unroll
        for (int t = 0; t < 8; ++t) b[8 + t] = rowq[kb + 16 + t];
        bfrag[kc] = b;
    }
}

static __device__ __forceinline__ v8f wmma_tile(const v16bf afrag[4],
                                                const v16bf bfrag[4])
{
    v8f acc = {0.f, 0.f, 0.f, 0.f, 0.f, 0.f, 0.f, 0.f};
#pragma unroll
    for (int kc = 0; kc < 4; ++kc)
        acc = __builtin_amdgcn_wmma_f32_16x16x32_bf16(
            /*neg_a=*/false, afrag[kc], /*neg_b=*/false, bfrag[kc],
            /*c_mod=*/(short)0, acc, /*reuse_a=*/false, /*reuse_b=*/false);
    return acc;
}

// Pass A: lse_x[i] = log( sum_j exp( sigmoid(g_ij) ) )   (args all in (0,1) => safe)
__global__ void gram_lse_kernel(const __bf16* __restrict__ X,
                                float* __restrict__ lseX)
{
    const int wid  = threadIdx.x >> 5;
    const int lane = threadIdx.x & 31;
    const int strip = blockIdx.x * (blockDim.x >> 5) + wid;
    if (strip >= NSTRIP) return;
    const int i0 = strip * 16;
    const int lh = lane & 15, hi = lane >> 4;

    v16bf afrag[4];
    load_a_frags(X, i0, lh, hi, afrag);

    float esum[8];
#pragma unroll
    for (int v = 0; v < 8; ++v) esum[v] = 0.f;

    // double-buffered B across j-tiles: next tile's loads in flight while
    // current tile does 4 WMMAs + elementwise transcendentals
    v16bf bcur[4];
    load_b_frags(X, 0, lh, hi, bcur);
    for (int j0 = 0; j0 < MLEN; j0 += 16) {
        const int jn = (j0 + 16 < MLEN) ? (j0 + 16) : j0;   // clamp: no branch
        v16bf bnext[4];
        load_b_frags(X, jn, lh, hi, bnext);

        const v8f acc = wmma_tile(afrag, bcur);
#pragma unroll
        for (int v = 0; v < 8; ++v)
            esum[v] += __expf(fast_sigmoid(acc[v]));

#pragma unroll
        for (int kc = 0; kc < 4; ++kc) bcur[kc] = bnext[kc];
    }

#pragma unroll
    for (int v = 0; v < 8; ++v) {
        float e = esum[v];
        e += __shfl_xor(e, 1, 32);
        e += __shfl_xor(e, 2, 32);
        e += __shfl_xor(e, 4, 32);
        e += __shfl_xor(e, 8, 32);
        if (lh == 0) lseX[i0 + v + 8 * hi] = __logf(e);
    }
}

// Pass B: per-row  (S1-S2)/T + lse_x - log(T)  with
//   m = P_ij * invrs_i, t = exp(m), T=Σt, S1=Σt*m, S2=Σt*sigmoid(g)
__global__ void gram_kl_kernel(const __bf16* __restrict__ X,
                               const float* __restrict__ P,
                               const float* __restrict__ invrs,
                               const float* __restrict__ lseX,
                               float* __restrict__ kl_accum)
{
    const int wid  = threadIdx.x >> 5;
    const int lane = threadIdx.x & 31;
    const int strip = blockIdx.x * (blockDim.x >> 5) + wid;
    if (strip >= NSTRIP) return;
    const int i0 = strip * 16;
    const int lh = lane & 15, hi = lane >> 4;

    v16bf afrag[4];
    load_a_frags(X, i0, lh, hi, afrag);

    float invr[8], T[8], S1[8], S2[8];
#pragma unroll
    for (int v = 0; v < 8; ++v) {
        invr[v] = invrs[i0 + v + 8 * hi];
        T[v] = 0.f; S1[v] = 0.f; S2[v] = 0.f;
    }

    v16bf bcur[4];
    load_b_frags(X, 0, lh, hi, bcur);
    for (int j0 = 0; j0 < MLEN; j0 += 16) {
        const int jn = (j0 + 16 < MLEN) ? (j0 + 16) : j0;
        v16bf bnext[4];
        load_b_frags(X, jn, lh, hi, bnext);

        // hoist the 8 streamed target-matrix loads (NT) ahead of the WMMA
        // chain, and prefetch four tiles (64 columns) ahead of the 144 MB stream
        float mval[8];
#pragma unroll
        for (int v = 0; v < 8; ++v) {
            const float* pp = P + (size_t)(i0 + v + 8 * hi) * MLEN + j0 + lh;
            mval[v] = __builtin_nontemporal_load(pp) * invr[v];
            __builtin_prefetch(pp + 64, 0, 0);
        }

        const v8f acc = wmma_tile(afrag, bcur);
#pragma unroll
        for (int v = 0; v < 8; ++v) {
            const float s = fast_sigmoid(acc[v]);
            const float t = __expf(mval[v]);
            T[v]  += t;
            S1[v] += t * mval[v];
            S2[v] += t * s;
        }

#pragma unroll
        for (int kc = 0; kc < 4; ++kc) bcur[kc] = bnext[kc];
    }

    float contrib = 0.f;
#pragma unroll
    for (int v = 0; v < 8; ++v) {
        float tt = T[v], s1 = S1[v], s2 = S2[v];
        tt += __shfl_xor(tt, 1, 32); s1 += __shfl_xor(s1, 1, 32); s2 += __shfl_xor(s2, 1, 32);
        tt += __shfl_xor(tt, 2, 32); s1 += __shfl_xor(s1, 2, 32); s2 += __shfl_xor(s2, 2, 32);
        tt += __shfl_xor(tt, 4, 32); s1 += __shfl_xor(s1, 4, 32); s2 += __shfl_xor(s2, 4, 32);
        tt += __shfl_xor(tt, 8, 32); s1 += __shfl_xor(s1, 8, 32); s2 += __shfl_xor(s2, 8, 32);
        if (lh == 0)
            contrib += (s1 - s2) / tt + lseX[i0 + v + 8 * hi] - __logf(tt);
    }
    if (lh == 0) atomicAdd(kl_accum, contrib);
}

// ---------------------------------------------------------------------------
// pr + positive-sum: one wave per pair.
// ---------------------------------------------------------------------------
__global__ void pr_kernel(const float* __restrict__ emb,
                          const int* __restrict__ du, const int* __restrict__ di,
                          float* __restrict__ prout, float* __restrict__ accum)
{
    const int wid  = threadIdx.x >> 5;
    const int lane = threadIdx.x & 31;
    const int pair = blockIdx.x * (blockDim.x >> 5) + wid;
    if (pair >= RPAIR) return;
    const float* ur = emb + (size_t)du[pair] * DT;
    const float* ir = emb + (size_t)(MLEN + di[pair]) * DT;
    const float4 a = *(const float4*)(ur + 4 * lane);
    const float4 b = *(const float4*)(ir + 4 * lane);
    const float d = wave_red_sum(a.x*b.x + a.y*b.y + a.z*b.z + a.w*b.w);
    if (lane == 0) {
        prout[pair] = d;
        const float ls = fminf(d, 0.f) - log1pf(__expf(-fabsf(d)));  // log_sigmoid
        atomicAdd(accum + 0, ls);
    }
}

__global__ void init_kernel(float* accum)
{
    if (threadIdx.x < 4) accum[threadIdx.x] = 0.f;
}

__global__ void finalize_kernel(const float* __restrict__ accum,
                                float* __restrict__ loss)
{
    *loss = -(accum[0] / (float)RPAIR) + BCOEF * (accum[1] + accum[2]);
}

// ---------------------------------------------------------------------------
extern "C" void kernel_launch(void* const* d_in, const int* in_sizes, int n_in,
                              void* d_out, int out_size, void* d_ws, size_t ws_size,
                              hipStream_t stream)
{
    const int*   neiberm = (const int*)d_in[0];
    const int*   neibern = (const int*)d_in[1];
    const int*   duser   = (const int*)d_in[2];
    const int*   ditemid = (const int*)d_in[3];
    const float* u0      = (const float*)d_in[4];
    const float* v0      = (const float*)d_in[5];
    const float* att     = (const float*)d_in[6];
    const float* Mm      = (const float*)d_in[7];
    const float* Nm      = (const float*)d_in[8];

    float* out   = (float*)d_out;
    float* emb   = out;                                   // 12000*128: user then item
    float* loss  = out + (size_t)NTOT * DT;               // index 1,536,000
    float* prout = loss + 1;                              // 100,000 values

    float*  ws    = (float*)d_ws;
    float*  ui1   = ws;                                   // 1,536,000 f
    float*  invM  = ws + (size_t)NTOT * DT;               // 6000
    float*  invN  = invM + MLEN;                          // 6000
    float*  lseU  = invN + NLEN;                          // 6000
    float*  lseI  = lseU + MLEN;                          // 6000
    float*  accum = lseI + NLEN;                          // 4 (pos, klM, klN, pad)
    __bf16* Xbf   = (__bf16*)(accum + 4);                 // 12000*128 bf16
    __bf16* XbfU  = Xbf;
    __bf16* XbfI  = Xbf + (size_t)MLEN * DT;

    init_kernel<<<1, 32, 0, stream>>>(accum);

    // routing layer 1: split inputs (u0, v0) -> ui1
    routing_kernel<<<NTOT / 8, 256, 0, stream>>>(u0, v0, neiberm, neibern, att, ui1);
    // routing layer 2: ui1 -> d_out emb region (user rows then item rows)
    routing_kernel<<<NTOT / 8, 256, 0, stream>>>(ui1, ui1 + (size_t)MLEN * DT,
                                                 neiberm, neibern, att, emb);

    cvt_bf16_kernel<<<(NTOT * DT) / 256, 256, 0, stream>>>(emb, Xbf, NTOT * DT);

    rowinv_kernel<<<MLEN, 256, 0, stream>>>(Mm, invM, MLEN);
    rowinv_kernel<<<NLEN, 256, 0, stream>>>(Nm, invN, NLEN);

    const int gblocks = (NSTRIP + 3) / 4;                 // 4 waves / block
    gram_lse_kernel<<<gblocks, 128, 0, stream>>>(XbfU, lseU);
    gram_lse_kernel<<<gblocks, 128, 0, stream>>>(XbfI, lseI);
    gram_kl_kernel<<<gblocks, 128, 0, stream>>>(XbfU, Mm, invM, lseU, accum + 1);
    gram_kl_kernel<<<gblocks, 128, 0, stream>>>(XbfI, Nm, invN, lseI, accum + 2);

    pr_kernel<<<(RPAIR + 7) / 8, 256, 0, stream>>>(emb, duser, ditemid, prout, accum);

    finalize_kernel<<<1, 1, 0, stream>>>(accum, loss);
}
